// RelationalNetwork_15788299780487
// MI455X (gfx1250) — matrix operand order repaired and checked by reading, not compiled, not run on hardware
//
#include <hip/hip_runtime.h>

// ---------------------------------------------------------------------------
// Relational Network, fused for CDNA5 (gfx1250, wave32, WMMA 16x16x32 f16).
//
// h0[n,i,j] = relu(Apre[n,j] + Bpre[n,i] + Cpre[n])   (layer-0 factorization)
// Layers 1..3: 256x256 fp16 GEMMs over 1.23M pair rows via v_wmma, with
//  - weights staged to LDS in 64KB N-halves (async-to-LDS when available)
//  - wave-private activation rows updated in place (no inter-layer barriers)
//  - final layer sum-pooled in-register -> LDS ds_add_f32 -> global atomics
// ---------------------------------------------------------------------------

typedef __attribute__((ext_vector_type(16))) _Float16 v16h;
typedef __attribute__((ext_vector_type(8)))  _Float16 v8h;
typedef __attribute__((ext_vector_type(8)))  float    v8f;
typedef __attribute__((ext_vector_type(4)))  int      v4i;

typedef __attribute__((address_space(1))) v4i as1_v4i;   // global int4
typedef __attribute__((address_space(3))) v4i as3_v4i;   // LDS int4

#define NB      32          // batch
#define CIN     24          // channels
#define O       196         // 14*14 cells
#define GH      256         // hidden width
#define HD      128         // code width
#define AS      28          // output classes
#define TILE_M  112         // pairs per workgroup; 38416 = 343 * 112
#define TILES   343
#define NWAVES  7
#define TPB     (NWAVES * 32)   // 224 threads

#define ACT_BYTES   (TILE_M * GH * 2)   // 57344: fp16 activations, in-place
#define WLDS_BYTES  (128 * GH * 2)      // 65536: one N-half of a weight matrix
#define SMEM_BYTES  (ACT_BYTES + WLDS_BYTES + 4 * GH * 4)  // +colsum+3 biases

#if __has_builtin(__builtin_amdgcn_global_load_async_to_lds_b128)
#define HAVE_ASYNC_LDS 1
#else
#define HAVE_ASYNC_LDS 0
#endif

__device__ __forceinline__ void wait_asynccnt0() {
#if __has_builtin(__builtin_amdgcn_s_wait_asynccnt)
  __builtin_amdgcn_s_wait_asynccnt(0);
#else
  asm volatile("s_wait_asynccnt 0x0" ::: "memory");
#endif
}

// Cooperative stage of 128KB of N-major fp16 weights into LDS.
__device__ __forceinline__ void stage_weights(const _Float16* __restrict__ Wt,
                                              _Float16* __restrict__ wlds,
                                              int tid) {
#if HAVE_ASYNC_LDS
  // GLOBAL_LOAD_ASYNC_TO_LDS_B128: no VGPR round-trip, tracked by ASYNCcnt.
  const char* gsrc = (const char*)Wt;
  char* ldst = (char*)wlds;
  for (int idx = tid; idx < 8192; idx += TPB) {
    __builtin_amdgcn_global_load_async_to_lds_b128(
        (as1_v4i*)(gsrc + (size_t)idx * 16),
        (as3_v4i*)(ldst + (size_t)idx * 16),
        0, 0);
  }
  wait_asynccnt0();
#else
  const float4* __restrict__ srcw = (const float4*)Wt;   // 8192 x 16B
  float4* __restrict__ dstw = (float4*)wlds;
  for (int idx = tid; idx < 8192; idx += TPB) dstw[idx] = srcw[idx];
#endif
}

// ---------------------------------------------------------------------------
// Prep: Apre[n][q][k] = feats[n,q] @ W0[0:26],  Bpre = feats @ W0[26:52]
// ---------------------------------------------------------------------------
__global__ void rn_prep_ab(const float* __restrict__ x,
                           const float* __restrict__ gw0,
                           float* __restrict__ Apre,
                           float* __restrict__ Bpre) {
  __shared__ float feats[CIN + 2];
  const int b = blockIdx.x;           // n*O + p
  const int n = b / O, p = b - n * O;
  const int t = threadIdx.x;          // 0..255 = output feature
  if (t < CIN) {
    feats[t] = x[((size_t)n * CIN + t) * O + p];
  } else if (t == CIN) {
    feats[CIN]     = -7.0f + (float)(p / 14) * (14.0f / 13.0f); // linspace(-7,7,14)
  } else if (t == CIN + 1) {
    feats[CIN + 1] = -7.0f + (float)(p % 14) * (14.0f / 13.0f);
  }
  __syncthreads();
  float a = 0.0f, bb = 0.0f;
#pragma unroll
  for (int c = 0; c < CIN + 2; ++c) {
    const float f = feats[c];
    a  += f * gw0[c * GH + t];                 // rows 0..25  -> feats[j] term
    bb += f * gw0[(CIN + 2 + c) * GH + t];     // rows 26..51 -> feats[i] term
  }
  const size_t off = ((size_t)n * O + p) * GH + t;
  Apre[off] = a;
  Bpre[off] = bb;
}

// Cpre[n][k] = code[n] @ W0[52:180] + b0
__global__ void rn_prep_c(const float* __restrict__ code,
                          const float* __restrict__ gw0,
                          const float* __restrict__ gb0,
                          float* __restrict__ Cpre) {
  __shared__ float cs[HD];
  const int n = blockIdx.x, t = threadIdx.x;
  if (t < HD) cs[t] = code[n * HD + t];
  __syncthreads();
  float a = gb0[t];
#pragma unroll 8
  for (int h = 0; h < HD; ++h) a += cs[h] * gw0[(2 * (CIN + 2) + h) * GH + t];
  Cpre[n * GH + t] = a;
}

// Transpose gw{1,2,3} (K-major, fp32) -> Wt{1,2,3} (N-major, fp16)
__global__ void rn_prep_wt(const float* __restrict__ g1,
                           const float* __restrict__ g2,
                           const float* __restrict__ g3,
                           _Float16* __restrict__ w1,
                           _Float16* __restrict__ w2,
                           _Float16* __restrict__ w3) {
  const int b = blockIdx.x;
  const int w = b / GH, nrow = b - w * GH;
  const int k = threadIdx.x;
  const float*    g = (w == 0) ? g1 : ((w == 1) ? g2 : g3);
  _Float16*       d = (w == 0) ? w1 : ((w == 1) ? w2 : w3);
  d[nrow * GH + k] = (_Float16)g[k * GH + nrow];
}

__global__ void rn_zero(float* __restrict__ r) {
  r[blockIdx.x * 256 + threadIdx.x] = 0.0f;
}

// ---------------------------------------------------------------------------
// One layer: act(112x256 fp16, LDS, in place) <- relu(act @ W + b).
// Each wave owns rows [16w, 16w+16): A frags read only own rows, results
// written only to own rows -> no activation barriers. Weights staged to LDS
// in two N-halves; the half-0 output is stashed packed-fp16 in registers
// until half-1 (which still reads the old activations) completes.
//   A frag (16x32): lane<16 -> M=lane,  K=k0..k0+15 contiguous; lane>=16: K+16
//   B frag (32x16): lane<16 -> N=lane, K=k0..k0+15 contiguous; lane>=16: K+16
// ---------------------------------------------------------------------------
template <bool FINAL>
__device__ __forceinline__ void rn_layer(_Float16* __restrict__ act,
                                         _Float16* __restrict__ wlds,
                                         const _Float16* __restrict__ Wt,
                                         const float* __restrict__ bias,
                                         float* __restrict__ colsum,
                                         int tid, int wave, int lane) {
  const int mrow = wave * 16 + (lane & 15);
  const int aoff = (lane < 16) ? 0 : 16;
  const int boff = (lane < 16) ? 0 : 16;
  const int ncol = lane & 15;
  v8h stash[8];

#pragma unroll 1
  for (int nh = 0; nh < 2; ++nh) {
    stage_weights(Wt + (size_t)nh * 128 * GH, wlds, tid);
    __syncthreads();

    v8f acc[8];
#pragma unroll
    for (int t = 0; t < 8; ++t)
#pragma unroll
      for (int e = 0; e < 8; ++e) acc[t][e] = 0.0f;

#pragma unroll 2
    for (int kb = 0; kb < 8; ++kb) {
      const int k0 = kb * 32;
      const v8h a0 = *(const v8h*)(act + mrow * GH + k0 + aoff);
      const v8h a1 = *(const v8h*)(act + mrow * GH + k0 + aoff + 8);
      const v16h a = __builtin_shufflevector(a0, a1, 0, 1, 2, 3, 4, 5, 6, 7,
                                             8, 9, 10, 11, 12, 13, 14, 15);
      v16h b[8];
#pragma unroll
      for (int nt = 0; nt < 8; ++nt)
        b[nt] = *(const v16h*)(wlds + (size_t)(nt * 16 + ncol) * GH + k0 + boff);
#pragma unroll
      for (int nt = 0; nt < 8; ++nt)
        acc[nt] = __builtin_amdgcn_wmma_f32_16x16x32_f16(
            false, a, false, b[nt], (short)0, acc[nt], false, false);
    }

    // Epilogue. C/D layout: VGPR r -> (M=r | M=8+r per lane half), N=lane%16.
    if (FINAL) {
#pragma unroll
      for (int nt = 0; nt < 8; ++nt) {
        const int col = nh * 128 + nt * 16 + ncol;
        const float bv = bias[col];
        float part = 0.0f;
#pragma unroll
        for (int r = 0; r < 8; ++r) part += fmaxf(acc[nt][r] + bv, 0.0f);
        atomicAdd(colsum + col, part);          // ds_add_f32
      }
    } else if (nh == 0) {
#pragma unroll
      for (int nt = 0; nt < 8; ++nt) {
        const float bv = bias[nt * 16 + ncol];
#pragma unroll
        for (int r = 0; r < 8; ++r)
          stash[nt][r] = (_Float16)fmaxf(acc[nt][r] + bv, 0.0f);
      }
    } else {
      const int mbase = wave * 16 + ((lane < 16) ? 0 : 8);
#pragma unroll
      for (int nt = 0; nt < 8; ++nt) {
        const int col0 = nt * 16 + ncol;
        const int col1 = 128 + col0;
        const float bv = bias[col1];
#pragma unroll
        for (int r = 0; r < 8; ++r) {
          act[(mbase + r) * GH + col0] = stash[nt][r];
          act[(mbase + r) * GH + col1] =
              (_Float16)fmaxf(acc[nt][r] + bv, 0.0f);
        }
      }
    }
    __syncthreads();   // all waves done with this wlds stage before overwrite
  }
}

__global__ __launch_bounds__(TPB) void rn_main(
    const float* __restrict__ Apre, const float* __restrict__ Bpre,
    const float* __restrict__ Cpre,
    const _Float16* __restrict__ Wt1, const float* __restrict__ gb1,
    const _Float16* __restrict__ Wt2, const float* __restrict__ gb2,
    const _Float16* __restrict__ Wt3, const float* __restrict__ gb3,
    float* __restrict__ relations) {
  extern __shared__ char smem_raw[];
  _Float16* act    = (_Float16*)smem_raw;                       // 112x256 fp16
  _Float16* wlds   = (_Float16*)(smem_raw + ACT_BYTES);         // 128x256 fp16
  float*    colsum = (float*)(smem_raw + ACT_BYTES + WLDS_BYTES);
  float*    bias1  = colsum + GH;
  float*    bias2  = bias1 + GH;
  float*    bias3  = bias2 + GH;

  const int n    = blockIdx.x / TILES;
  const int tile = blockIdx.x - n * TILES;
  const int tid  = threadIdx.x;
  const int wave = tid >> 5;
  const int lane = tid & 31;

  for (int c = tid; c < GH; c += TPB) {
    colsum[c] = 0.0f;
    bias1[c] = gb1[c];
    bias2[c] = gb2[c];
    bias3[c] = gb3[c];
  }

  // Stage h0 = relu(Apre[n,j] + Bpre[n,i] + Cpre[n]) -> act (fp16).
  // Rows [16w,16w+16) staged by wave w's threads -> wave-private, no barrier.
  {
    const int row = tid >> 1;
    const int cb  = (tid & 1) * 128;
    const int p   = tile * TILE_M + row;        // pair index: p = i*O + j
    const int i   = p / O;
    const int j   = p - i * O;
    const float* ap = Apre + ((size_t)n * O + j) * GH + cb;
    const float* bp = Bpre + ((size_t)n * O + i) * GH + cb;
    const float* cp = Cpre + (size_t)n * GH + cb;
    _Float16* dstp  = act + row * GH + cb;
    for (int k = 0; k < 128; k += 8) {
      v8h h;
#pragma unroll
      for (int e = 0; e < 8; ++e) {
        const float v = ap[k + e] + bp[k + e] + cp[k + e];
        h[e] = (_Float16)fmaxf(v, 0.0f);
      }
      *(v8h*)(dstp + k) = h;
    }
  }

  rn_layer<false>(act, wlds, Wt1, bias1, nullptr, tid, wave, lane);
  rn_layer<false>(act, wlds, Wt2, bias2, nullptr, tid, wave, lane);
  rn_layer<true >(act, wlds, Wt3, bias3, colsum, tid, wave, lane);

  for (int c = tid; c < GH; c += TPB)
    atomicAdd(relations + (size_t)n * GH + c, colsum[c]);
}

// ---------------------------------------------------------------------------
// f-head: relations(32x256) -> relu@fw0 -> relu@fw1 -> @fw2 (32x28). Tiny.
// ---------------------------------------------------------------------------
__global__ void rn_fnet(const float* __restrict__ relations,
                        const float* __restrict__ fw0, const float* __restrict__ fb0,
                        const float* __restrict__ fw1, const float* __restrict__ fb1,
                        const float* __restrict__ fw2, const float* __restrict__ fb2,
                        float* __restrict__ out) {
  __shared__ float s0[GH], s1[GH];
  const int n = blockIdx.x, t = threadIdx.x;
  s0[t] = relations[n * GH + t];
  __syncthreads();
  float a = fb0[t];
#pragma unroll 8
  for (int k = 0; k < GH; ++k) a += s0[k] * fw0[k * GH + t];
  s1[t] = fmaxf(a, 0.0f);
  __syncthreads();
  a = fb1[t];
#pragma unroll 8
  for (int k = 0; k < GH; ++k) a += s1[k] * fw1[k * GH + t];
  __syncthreads();                 // all reads of s0 (layer0) are done
  s0[t] = fmaxf(a, 0.0f);
  __syncthreads();
  if (t < AS) {
    float o = fb2[t];
#pragma unroll 8
    for (int k = 0; k < GH; ++k) o += s0[k] * fw2[k * AS + t];
    out[n * AS + t] = o;
  }
}

// ---------------------------------------------------------------------------
extern "C" void kernel_launch(void* const* d_in, const int* in_sizes, int n_in,
                              void* d_out, int out_size, void* d_ws, size_t ws_size,
                              hipStream_t stream) {
  (void)in_sizes; (void)n_in; (void)out_size; (void)ws_size;
  const float* x    = (const float*)d_in[0];
  const float* code = (const float*)d_in[1];
  const float* gw0  = (const float*)d_in[2];
  const float* gb0  = (const float*)d_in[3];
  const float* gw1  = (const float*)d_in[4];
  const float* gb1  = (const float*)d_in[5];
  const float* gw2  = (const float*)d_in[6];
  const float* gb2  = (const float*)d_in[7];
  const float* gw3  = (const float*)d_in[8];
  const float* gb3  = (const float*)d_in[9];
  const float* fw0  = (const float*)d_in[10];
  const float* fb0  = (const float*)d_in[11];
  const float* fw1  = (const float*)d_in[12];
  const float* fb1  = (const float*)d_in[13];
  const float* fw2  = (const float*)d_in[14];
  const float* fb2  = (const float*)d_in[15];

  // Workspace carve-out (~12.7 MB total).
  float*    Apre = (float*)d_ws;                    // 32*196*256 f32
  float*    Bpre = Apre + (size_t)NB * O * GH;      // 32*196*256 f32
  float*    Cpre = Bpre + (size_t)NB * O * GH;      // 32*256 f32
  _Float16* Wt1  = (_Float16*)(Cpre + (size_t)NB * GH);
  _Float16* Wt2  = Wt1 + (size_t)GH * GH;
  _Float16* Wt3  = Wt2 + (size_t)GH * GH;
  float* relations = (float*)(Wt3 + (size_t)GH * GH); // 32*256 f32
  float* out = (float*)d_out;

  rn_prep_ab<<<NB * O, 256, 0, stream>>>(x, gw0, Apre, Bpre);
  rn_prep_c<<<NB, 256, 0, stream>>>(code, gw0, gb0, Cpre);
  rn_prep_wt<<<3 * GH, 256, 0, stream>>>(gw1, gw2, gw3, Wt1, Wt2, Wt3);
  rn_zero<<<NB, 256, 0, stream>>>(relations);
  rn_main<<<NB * TILES, TPB, SMEM_BYTES, stream>>>(
      Apre, Bpre, Cpre, Wt1, gb1, Wt2, gb2, Wt3, gb3, relations);
  rn_fnet<<<NB, 256, 0, stream>>>(relations, fw0, fb0, fw1, fb1, fw2, fb2, out);
}
